// NodeEmbeddingGraphSage_13511967113599
// MI455X (gfx1250) — matrix operand order repaired and checked by reading, not compiled, not run on hardware
//
#include <hip/hip_runtime.h>

#define N_NODES 100000
#define DIM 128
#define N_REL 3
#define E_REL 200000

typedef __attribute__((ext_vector_type(16))) _Float16 v16h;
typedef __attribute__((ext_vector_type(8)))  _Float16 v8h;
typedef __attribute__((ext_vector_type(8)))  float    v8f;

// ---------------- elementwise helpers ----------------

__global__ void zero_f32_kernel(float* __restrict__ p, int n) {
    int i = blockIdx.x * blockDim.x + threadIdx.x;
    if (i < n) p[i] = 0.0f;
}

// ---------------- W pre-pack: f32 [K][N] -> f16 in WMMA-B register order ----
// Dest layout: [kk2 = K/32][hi = (K/16)&1][N][i = K&15] f16, so a lane's 16
// B values (fixed N, K = kk + hi*16 .. +15) are 32 contiguous bytes.
__global__ void pack_w_kernel(const float* __restrict__ W, _Float16* __restrict__ Wp) {
    int gid = blockIdx.x * blockDim.x + threadIdx.x;   // over DIM*DIM
    if (gid < DIM * DIM) {
        int k = gid >> 7;
        int n = gid & (DIM - 1);
        int kk2 = k >> 5;
        int hi  = (k >> 4) & 1;
        int i   = k & 15;
        Wp[(((kk2 * 2 + hi) * DIM) + n) * 16 + i] = (_Float16)W[gid];
    }
}

// ---------------- aggregation (scatter) ----------------

__global__ void deg_kernel(const int* __restrict__ dst, float* __restrict__ deg) {
    int e = blockIdx.x * blockDim.x + threadIdx.x;
    if (e < E_REL) atomicAdd(&deg[dst[e]], 1.0f);
}

// 32 threads per edge, 4 floats per thread: msg[dst] += h[src]
__global__ void scatter_kernel(const float* __restrict__ h, const int* __restrict__ src,
                               const int* __restrict__ dst, float* __restrict__ msg) {
    int gid = blockIdx.x * blockDim.x + threadIdx.x;
    int e = gid >> 5;
    int j = (gid & 31) * 4;
    if (e < E_REL) {
        int s = src[e];
        int d = dst[e];
        const float4 v = *(const float4*)(h + (size_t)s * DIM + j);
        float* o = msg + (size_t)d * DIM + j;
        atomicAdd(o + 0, v.x);
        atomicAdd(o + 1, v.y);
        atomicAdd(o + 2, v.z);
        atomicAdd(o + 3, v.w);
    }
}

// normalize + convert once: agg16[n][j] = f16(msg[n][j] / max(deg[n],1))
__global__ void norm_kernel(const float* __restrict__ msg, const float* __restrict__ deg,
                            _Float16* __restrict__ agg16, int n) {
    int i = blockIdx.x * blockDim.x + threadIdx.x;
    if (i < n) {
        int row = i >> 7;
        float inv = 1.0f / fmaxf(deg[row], 1.0f);
        agg16[i] = (_Float16)(msg[i] * inv);
    }
}

// ---------------- fused layer GEMM: hout = relu(sum_r agg_r @ W_r + b) -----
// Block = 256 threads (8 waves) covers a 32(M) x 128(N) output tile.
// Wave w: M-tile = w>>2, N columns [(w&3)*32, +32) -> two 16x16 C tiles
// sharing the A registers. K = 3 relations x 128 = 12 k-steps of 32:
// 24 WMMAs per wave, C stays in VGPRs; bias + relu fused into the store.
__global__ void __launch_bounds__(256)
gemm_layer_kernel(const _Float16* __restrict__ agg,   // [N_REL][N][DIM] f16
                  const _Float16* __restrict__ Wp,    // [N_REL][DIM*DIM] f16 packed
                  const float* __restrict__ bias,     // [N_REL][DIM] f32
                  float* __restrict__ hout) {         // [N][DIM] f32
    const int lane = threadIdx.x & 31;
    const int wave = threadIdx.x >> 5;
    const int l15  = lane & 15;
    const int hi   = lane >> 4;                 // 0: lanes 0-15, 1: lanes 16-31
    const int m0   = blockIdx.x * 32 + (wave >> 2) * 16;
    const int n0   = (wave & 3) * 32;

    const int M = m0 + l15;
    const int aKoff = hi * 16;                  // A K-chunk base (contiguous K per lane half)

    v8f c0, c1;
#pragma unroll
    for (int i = 0; i < 8; ++i) { c0[i] = 0.0f; c1[i] = 0.0f; }

#pragma unroll
    for (int r = 0; r < N_REL; ++r) {
        const _Float16* arow  = agg + ((size_t)r * N_NODES + M) * DIM;
        const _Float16* wbase = Wp + (size_t)r * DIM * DIM;
#pragma unroll
        for (int kk = 0; kk < DIM; kk += 32) {
            // A (16x32 f16): lanes 0-15 -> K {0..15}; lanes 16-31 -> K {16..31}
            v8h alo = *(const v8h*)(arow + kk + aKoff);
            v8h ahi = *(const v8h*)(arow + kk + aKoff + 8);
            v16h a;
#pragma unroll
            for (int i = 0; i < 8; ++i) { a[i] = alo[i]; a[8 + i] = ahi[i]; }

            // B: packed layout -> one contiguous 32B chunk per lane per tile
            const _Float16* bbase = wbase + (size_t)((kk >> 5) * 2 + hi) * DIM * 16;
            v16h b0 = *(const v16h*)(bbase + (size_t)(n0 + l15) * 16);
            v16h b1 = *(const v16h*)(bbase + (size_t)(n0 + 16 + l15) * 16);

            c0 = __builtin_amdgcn_wmma_f32_16x16x32_f16(false, a, false, b0,
                                                        (short)0, c0, false, false);
            c1 = __builtin_amdgcn_wmma_f32_16x16x32_f16(false, a, false, b1,
                                                        (short)0, c1, false, false);
        }
    }

    // bias (summed over relations) + relu, direct store (no RMW)
    const int nc0 = n0 + l15;
    const int nc1 = nc0 + 16;
    const float bs0 = bias[nc0] + bias[DIM + nc0] + bias[2 * DIM + nc0];
    const float bs1 = bias[nc1] + bias[DIM + nc1] + bias[2 * DIM + nc1];

    // C/D layout: VGPR v -> row m0 + v + hi*8, column n0 (+16) + l15
    float* o0 = hout + (size_t)(m0 + hi * 8) * DIM + nc0;
    float* o1 = o0 + 16;
#pragma unroll
    for (int v = 0; v < 8; ++v) {
        o0[(size_t)v * DIM] = fmaxf(c0[v] + bs0, 0.0f);
        o1[(size_t)v * DIM] = fmaxf(c1[v] + bs1, 0.0f);
    }
}

// ---------------- driver ----------------

extern "C" void kernel_launch(void* const* d_in, const int* in_sizes, int n_in,
                              void* d_out, int out_size, void* d_ws, size_t ws_size,
                              hipStream_t stream) {
    (void)in_sizes; (void)n_in; (void)out_size; (void)ws_size;

    const float* x   = (const float*)d_in[0];
    const int*   src = (const int*)d_in[1];    // [R][E]
    const int*   dst = (const int*)d_in[2];    // [R][E]
    const float* Wl[3] = { (const float*)d_in[3], (const float*)d_in[5], (const float*)d_in[7] };
    const float* Bl[3] = { (const float*)d_in[4], (const float*)d_in[6], (const float*)d_in[8] };
    float* out = (float*)d_out;

    const size_t ND = (size_t)N_NODES * DIM;
    float*    msg   = (float*)d_ws;                       // [N][DIM] f32
    float*    deg   = msg + ND;                           // [N]      f32
    float*    h1    = deg + N_NODES;                      // [N][DIM] f32
    _Float16* agg16 = (_Float16*)(h1 + ND);               // [R][N][DIM] f16
    _Float16* Wp    = agg16 + (size_t)N_REL * ND;         // [9][DIM*DIM] f16 packed

    // layer activations: L1 in=x out=h1; L2 in=h1 out=d_out; L3 in=d_out out=d_out
    const float* hin[3]  = { x, h1, out };
    float*       hout[3] = { h1, out, out };

    const int T = 256;
    const int nd_blocks = (int)((ND + T - 1) / T);
    const int n_blocks  = (N_NODES + T - 1) / T;
    const int e_blocks  = (E_REL + T - 1) / T;
    const int se_blocks = (int)(((size_t)E_REL * 32 + T - 1) / T);
    const int w_blocks  = (DIM * DIM + T - 1) / T;
    const int m_blocks  = N_NODES / 32;      // 3125, exact

    // Pre-pack all 9 weight matrices into WMMA-B f16 layout (every call; stateless)
    for (int l = 0; l < 3; ++l)
        for (int r = 0; r < N_REL; ++r)
            pack_w_kernel<<<w_blocks, T, 0, stream>>>(
                Wl[l] + (size_t)r * DIM * DIM, Wp + (size_t)(l * N_REL + r) * DIM * DIM);

    for (int l = 0; l < 3; ++l) {
        // aggregate all 3 relations into agg16[r]
        for (int r = 0; r < N_REL; ++r) {
            zero_f32_kernel<<<nd_blocks, T, 0, stream>>>(msg, (int)ND);
            zero_f32_kernel<<<n_blocks,  T, 0, stream>>>(deg, N_NODES);
            deg_kernel<<<e_blocks, T, 0, stream>>>(dst + (size_t)r * E_REL, deg);
            scatter_kernel<<<se_blocks, T, 0, stream>>>(hin[l], src + (size_t)r * E_REL,
                                                        dst + (size_t)r * E_REL, msg);
            norm_kernel<<<nd_blocks, T, 0, stream>>>(msg, deg, agg16 + (size_t)r * ND, (int)ND);
        }
        // one fused K=384 GEMM: hout = relu(sum_r agg_r @ W_r + sum_r b_r)
        gemm_layer_kernel<<<m_blocks, T, 0, stream>>>(
            agg16, Wp + (size_t)l * N_REL * DIM * DIM, Bl[l], hout[l]);
    }
}
